// GCNMae_76192719832100
// MI455X (gfx1250) — matrix operand order repaired and checked by reading, not compiled, not run on hardware
//
#include <hip/hip_runtime.h>
#include <hip/hip_bf16.h>

// ---------------- problem constants ----------------
constexpr int NG   = 256;           // graphs (batch)
constexpr int NV   = 68;            // nodes per graph
constexpr int EPER = 1088;          // edges per graph
constexpr int NE   = NG * EPER;     // 278528 total edges
constexpr int NN   = NG * NV;       // 17408 total nodes
constexpr int C1   = 1024;
constexpr int C2   = 512;
constexpr int C3   = 256;
constexpr int LATD = 128;
constexpr int KTOP = 34;
constexpr int KC   = KTOP * C3;     // 8704

// ---------------- WMMA types ----------------
typedef __attribute__((ext_vector_type(16))) _Float16 v16h;
typedef __attribute__((ext_vector_type(8)))  _Float16 v8h;
typedef __attribute__((ext_vector_type(8)))  float    v8f;
typedef __attribute__((ext_vector_type(4)))  int      v4i_;

// ---------------- CDNA5 async global->LDS copy (guarded) ----------------
#if defined(__has_builtin)
#if __has_builtin(__builtin_amdgcn_global_load_async_to_lds_b128)
#define HAVE_ASYNC_LDS 1
#endif
#endif

#ifdef HAVE_ASYNC_LDS
#if __has_builtin(__builtin_amdgcn_s_wait_asynccnt)
#define WAIT_ASYNC0() __builtin_amdgcn_s_wait_asynccnt(0)
#else
#define WAIT_ASYNC0() asm volatile("s_wait_asynccnt 0x0" ::: "memory")
#endif
#else
#define WAIT_ASYNC0() ((void)0)
#endif

// =====================================================================
// Graph preprocessing: degree, dinv, dense normalized adjacency
// =====================================================================
__global__ void gcn_deg_kernel(const int* __restrict__ ei,
                               const float* __restrict__ ew,
                               float* __restrict__ deg) {
  int e = blockIdx.x * blockDim.x + threadIdx.x;
  if (e >= NE) return;
  int g = e / EPER;
  int d = ei[NE + e];
  atomicAdd(&deg[g * NV + d], ew[e]);
}

__global__ void gcn_dinv_kernel(const float* __restrict__ deg,
                                float* __restrict__ dinv) {
  int i = blockIdx.x * blockDim.x + threadIdx.x;
  if (i < NN) dinv[i] = rsqrtf(deg[i] + 1.0f);   // self-loop weight 1
}

__global__ void gcn_adj_kernel(const int* __restrict__ ei,
                               const float* __restrict__ ew,
                               const float* __restrict__ dinv,
                               float* __restrict__ A) {
  int e = blockIdx.x * blockDim.x + threadIdx.x;
  if (e >= NE) return;
  int g = e / EPER;
  int s = ei[e];
  int d = ei[NE + e];
  float w = dinv[g * NV + s] * ew[e] * dinv[g * NV + d];
  atomicAdd(&A[(size_t)g * NV * NV + d * NV + s], w);
}

__global__ void gcn_adj_diag_kernel(const float* __restrict__ dinv,
                                    float* __restrict__ A) {
  int i = blockIdx.x * blockDim.x + threadIdx.x;
  if (i >= NN) return;
  int g = i / NV, v = i % NV;
  float di = dinv[i];
  A[(size_t)g * NV * NV + v * NV + v] += di * di;
}

// =====================================================================
// Weight prep: W[K][Nc] f32  ->  WT[Nc][K] f16 (col-major for B staging)
// =====================================================================
__global__ void gcn_wtT_kernel(const float* __restrict__ W,
                               _Float16* __restrict__ WT, int K, int Ncol) {
  size_t idx = (size_t)blockIdx.x * blockDim.x + threadIdx.x;
  if (idx >= (size_t)K * Ncol) return;
  int k = (int)(idx / Ncol), c = (int)(idx % Ncol);
  WT[(size_t)c * K + k] = (_Float16)W[idx];
}

// =====================================================================
// conv1 (D=1 collapse): s1 = A @ x ;  h1 = relu(s1 * W1 + b1) (f16 out)
// =====================================================================
__global__ void gcn_conv1_s_kernel(const float* __restrict__ A,
                                   const float* __restrict__ x,
                                   float* __restrict__ s1) {
  int g = blockIdx.x, t = threadIdx.x;
  __shared__ float xs[NV];
  if (t < NV) xs[t] = x[g * NV + t];
  __syncthreads();
  if (t < NV) {
    const float* arow = A + (size_t)g * NV * NV + t * NV;
    float acc = 0.f;
    #pragma unroll 4
    for (int u = 0; u < NV; ++u) acc += arow[u] * xs[u];
    s1[g * NV + t] = acc;
  }
}

__global__ void gcn_h1_kernel(const float* __restrict__ s1,
                              const float* __restrict__ W1,
                              const float* __restrict__ b1,
                              _Float16* __restrict__ h1) {
  size_t idx = (size_t)blockIdx.x * blockDim.x + threadIdx.x;
  if (idx >= (size_t)NN * C1) return;
  int i = (int)(idx / C1), c = (int)(idx % C1);
  h1[idx] = (_Float16)fmaxf(s1[i] * W1[c] + b1[c], 0.f);
}

// =====================================================================
// WMMA tiled GEMM: Cop[M,Nc] = A[M,K](f16) @ BT[Nc,K](f16)^T (+bias)
// 256 threads = 8 waves; block tile 128x128; wave tile 32x64 (2x4 WMMA)
// Double-buffered LDS staged with async global->LDS DMA when available.
// Requires M%128==0, Nc%128==0, K%32==0.
// =====================================================================
#define GBM 128
#define GBN 128
#define GBK 32

__device__ __forceinline__ void gcn_copy16_g2l(const _Float16* g,
                                               _Float16* l) {
#ifdef HAVE_ASYNC_LDS
  __builtin_amdgcn_global_load_async_to_lds_b128(
      (__attribute__((address_space(1))) v4i_*)(_Float16*)g,
      (__attribute__((address_space(3))) v4i_*)l, 0, 0);
#else
  *(uint4*)l = *(const uint4*)g;
#endif
}

__device__ __forceinline__ void gcn_stage(const _Float16* __restrict__ Abase,
                                          const _Float16* __restrict__ BTbase,
                                          _Float16* As, _Float16* Bs,
                                          int tid, int bm, int bn,
                                          int Kd, int k0) {
  const int row0 = tid >> 2;             // 0..63
  const int ch0  = (tid & 3) << 3;       // 0,8,16,24 (halfs)
#ifdef HAVE_ASYNC_LDS
  gcn_copy16_g2l(Abase + (size_t)(bm + row0) * Kd + k0 + ch0,
                 As + row0 * GBK + ch0);
  gcn_copy16_g2l(Abase + (size_t)(bm + row0 + 64) * Kd + k0 + ch0,
                 As + (row0 + 64) * GBK + ch0);
  gcn_copy16_g2l(BTbase + (size_t)(bn + row0) * Kd + k0 + ch0,
                 Bs + row0 * GBK + ch0);
  gcn_copy16_g2l(BTbase + (size_t)(bn + row0 + 64) * Kd + k0 + ch0,
                 Bs + (row0 + 64) * GBK + ch0);
#else
  // batched loads first (keep them all in flight), then LDS stores
  uint4 a0 = *(const uint4*)(Abase + (size_t)(bm + row0) * Kd + k0 + ch0);
  uint4 a1 = *(const uint4*)(Abase + (size_t)(bm + row0 + 64) * Kd + k0 + ch0);
  uint4 b0 = *(const uint4*)(BTbase + (size_t)(bn + row0) * Kd + k0 + ch0);
  uint4 b1 = *(const uint4*)(BTbase + (size_t)(bn + row0 + 64) * Kd + k0 + ch0);
  *(uint4*)(As + row0 * GBK + ch0) = a0;
  *(uint4*)(As + (row0 + 64) * GBK + ch0) = a1;
  *(uint4*)(Bs + row0 * GBK + ch0) = b0;
  *(uint4*)(Bs + (row0 + 64) * GBK + ch0) = b1;
#endif
}

__global__ __launch_bounds__(256)
void gemm_wmma_kernel(const _Float16* __restrict__ Aop,
                      const _Float16* __restrict__ BTop,
                      const float* __restrict__ bias,
                      float* __restrict__ Cop,
                      int M, int Kd, int Nc, int fuse) {
  __shared__ alignas(32) _Float16 As[2][GBM * GBK];  // row-major [m][k]
  __shared__ alignas(32) _Float16 Bs[2][GBN * GBK];  // col-major [n][k]

  const int tid  = threadIdx.x;
  const int bm   = blockIdx.y * GBM;
  const int bn   = blockIdx.x * GBN;
  const int wid  = tid >> 5;
  const int lane = tid & 31;
  const int wm   = (wid >> 1) * 32;   // 0,32,64,96
  const int wn   = (wid & 1) * 64;    // 0,64
  const int g16  = lane >> 4;
  const int l16  = lane & 15;

  v8f acc[2][4];
  #pragma unroll
  for (int i = 0; i < 2; ++i)
    #pragma unroll
    for (int j = 0; j < 4; ++j)
      acc[i][j] = (v8f){0.f, 0.f, 0.f, 0.f, 0.f, 0.f, 0.f, 0.f};

  const int nk = Kd / GBK;
  gcn_stage(Aop, BTop, As[0], Bs[0], tid, bm, bn, Kd, 0);   // prologue tile 0

  for (int kt = 0; kt < nk; ++kt) {
    WAIT_ASYNC0();          // this wave's copies of tile kt are done
    __syncthreads();        // tile kt visible to all waves
    if (kt + 1 < nk)        // prefetch next tile into the other buffer
      gcn_stage(Aop, BTop, As[(kt + 1) & 1], Bs[(kt + 1) & 1],
                tid, bm, bn, Kd, (kt + 1) * GBK);

    const _Float16* Asb = As[kt & 1];
    const _Float16* Bsb = Bs[kt & 1];

    // ---- fragments per documented wave32 layouts ----
    v16h afrag[2], bfrag[4];
    #pragma unroll
    for (int i = 0; i < 2; ++i) {
      int row = wm + i * 16 + l16;
      v8h lo = *(const v8h*)&Asb[row * GBK + g16 * 8];        // k in [8g, 8g+8)
      v8h hi = *(const v8h*)&Asb[row * GBK + 16 + g16 * 8];   // k in [16+8g, 24+8g)
      afrag[i] = __builtin_shufflevector(lo, hi, 0, 1, 2, 3, 4, 5, 6, 7,
                                         8, 9, 10, 11, 12, 13, 14, 15);
    }
    #pragma unroll
    for (int j = 0; j < 4; ++j) {
      int col = wn + j * 16 + l16;
      v8h blo = *(const v8h*)&Bsb[col * GBK + g16 * 8];       // k in [8g, 8g+8)
      v8h bhi = *(const v8h*)&Bsb[col * GBK + 16 + g16 * 8];  // k in [16+8g, 24+8g)
      bfrag[j] = __builtin_shufflevector(blo, bhi, 0, 1, 2, 3, 4, 5, 6, 7,
                                         8, 9, 10, 11, 12, 13, 14, 15);
    }

    #pragma unroll
    for (int i = 0; i < 2; ++i)
      #pragma unroll
      for (int j = 0; j < 4; ++j)
        acc[i][j] = __builtin_amdgcn_wmma_f32_16x16x32_f16(
            false, afrag[i], false, bfrag[j], (short)0, acc[i][j], false, false);

    __syncthreads();        // all waves done reading buf[kt&1]
  }

  // ---- epilogue: D layout m = r + 8*g16, n = l16 ----
  #pragma unroll
  for (int i = 0; i < 2; ++i) {
    #pragma unroll
    for (int j = 0; j < 4; ++j) {
      int coln = bn + wn + j * 16 + l16;
      float bv = (fuse >= 1) ? bias[coln] : 0.f;
      #pragma unroll
      for (int r = 0; r < 8; ++r) {
        int rowm = bm + wm + i * 16 + g16 * 8 + r;
        float v = acc[i][j][r] + bv;
        if (fuse == 2) v = fmaxf(v, 0.f);
        Cop[(size_t)rowm * Nc + coln] = v;
      }
    }
  }
}

// =====================================================================
// Aggregation: out[g*NV+v, c] = relu( sum_u A[g,v,u]*xw[g*NV+u, c] + b[c] )
// OutT selects f16 (feeds next WMMA GEMM) or f32 (feeds topk)
// =====================================================================
template <typename OutT>
__global__ __launch_bounds__(256)
void gcn_agg_kernel(const float* __restrict__ A,
                    const float* __restrict__ xw,
                    const float* __restrict__ bias,
                    OutT* __restrict__ out, int C) {
  int g = blockIdx.x;
  __shared__ float Ash[NV * NV];
  const float* Ag = A + (size_t)g * NV * NV;
  for (int i = threadIdx.x; i < NV * NV; i += blockDim.x) Ash[i] = Ag[i];
  __syncthreads();
  int total = NV * C;
  for (int idx = threadIdx.x; idx < total; idx += blockDim.x) {
    int v = idx / C, c = idx % C;
    const float* arow = Ash + v * NV;
    const float* xcol = xw + (size_t)g * NV * C + c;
    float acc = 0.f;
    #pragma unroll 4
    for (int u = 0; u < NV; ++u) acc += arow[u] * xcol[(size_t)u * C];
    out[(size_t)(g * NV + v) * C + c] = (OutT)fmaxf(acc + bias[c], 0.f);
  }
}

// =====================================================================
// TopK pooling: score = tanh(h3 . w / ||w||); keep top-34 (stable desc),
// z[g, rank*C3 + c] = h3[sel, c] * score[sel]   (f16 out, feeds head GEMM)
// =====================================================================
__global__ void gcn_topk_kernel(const float* __restrict__ h3,
                                const float* __restrict__ pw,
                                _Float16* __restrict__ z) {
  int g = blockIdx.x, t = threadIdx.x;   // 128 threads
  __shared__ float red[128];
  __shared__ float sc[NV];
  __shared__ float wn;
  float p = 0.f;
  for (int c = t; c < C3; c += 128) p += pw[c] * pw[c];
  red[t] = p; __syncthreads();
  for (int s = 64; s > 0; s >>= 1) { if (t < s) red[t] += red[t + s]; __syncthreads(); }
  if (t == 0) wn = sqrtf(red[0]);
  __syncthreads();
  if (t < NV) {
    const float* hr = h3 + (size_t)(g * NV + t) * C3;
    float acc = 0.f;
    for (int c = 0; c < C3; ++c) acc += hr[c] * pw[c];
    sc[t] = tanhf(acc / wn);
  }
  __syncthreads();
  if (t < NV) {
    float my = sc[t];
    int rank = 0;
    for (int u = 0; u < NV; ++u) {
      float o = sc[u];
      if (o > my || (o == my && u < t)) rank++;
    }
    if (rank < KTOP) {
      const float* hr = h3 + (size_t)(g * NV + t) * C3;
      _Float16* zr = z + (size_t)g * KC + (size_t)rank * C3;
      for (int c = 0; c < C3; ++c) zr[c] = (_Float16)(hr[c] * my);
    }
  }
}

// =====================================================================
// LayerNorm over 128 (in place, f32)
// =====================================================================
__global__ void gcn_ln128_kernel(float* __restrict__ z,
                                 const float* __restrict__ w,
                                 const float* __restrict__ b) {
  int r = blockIdx.x, t = threadIdx.x;   // 128 threads
  __shared__ float red[128];
  float v = z[(size_t)r * LATD + t];
  red[t] = v; __syncthreads();
  for (int s = 64; s > 0; s >>= 1) { if (t < s) red[t] += red[t + s]; __syncthreads(); }
  float m = red[0] / (float)LATD;
  __syncthreads();
  float d = v - m;
  red[t] = d * d; __syncthreads();
  for (int s = 64; s > 0; s >>= 1) { if (t < s) red[t] += red[t + s]; __syncthreads(); }
  float ri = rsqrtf(red[0] / (float)LATD + 1e-5f);
  z[(size_t)r * LATD + t] = d * ri * w[t] + b[t];
}

// =====================================================================
// Decoder row: z2[g,v] = LN68( relu( zlat[g,:] @ dec_W[:,v] + dec_b[v] ) )
// =====================================================================
__global__ void gcn_dec_kernel(const float* __restrict__ zlat,
                               const float* __restrict__ decW,
                               const float* __restrict__ decb,
                               const float* __restrict__ dnw,
                               const float* __restrict__ dnb,
                               float* __restrict__ z2) {
  int g = blockIdx.x, t = threadIdx.x;   // 128 threads
  __shared__ float zl[LATD];
  __shared__ float vals[NV];
  __shared__ float st[2];
  if (t < LATD) zl[t] = zlat[(size_t)g * LATD + t];
  __syncthreads();
  float val = 0.f;
  if (t < NV) {
    float acc = decb[t];
    for (int k = 0; k < LATD; ++k) acc += zl[k] * decW[(size_t)k * NV + t];
    val = fmaxf(acc, 0.f);
    vals[t] = val;
  }
  __syncthreads();
  if (t == 0) {
    float m = 0.f;
    for (int u = 0; u < NV; ++u) m += vals[u];
    m /= (float)NV;
    float var = 0.f;
    for (int u = 0; u < NV; ++u) { float d = vals[u] - m; var += d * d; }
    var /= (float)NV;
    st[0] = m; st[1] = rsqrtf(var + 1e-5f);
  }
  __syncthreads();
  if (t < NV) z2[g * NV + t] = (val - st[0]) * st[1] * dnw[t] + dnb[t];
}

// =====================================================================
// Final conv (D=1): out[g,v] = relu( (A @ z2)[g,v] * w + b )
// =====================================================================
__global__ void gcn_dconv_kernel(const float* __restrict__ A,
                                 const float* __restrict__ z2,
                                 const float* __restrict__ w,
                                 const float* __restrict__ b,
                                 float* __restrict__ out) {
  int g = blockIdx.x, t = threadIdx.x;
  __shared__ float zs[NV];
  if (t < NV) zs[t] = z2[g * NV + t];
  __syncthreads();
  if (t < NV) {
    const float* arow = A + (size_t)g * NV * NV + t * NV;
    float acc = 0.f;
    #pragma unroll 4
    for (int u = 0; u < NV; ++u) acc += arow[u] * zs[u];
    out[g * NV + t] = fmaxf(acc * w[0] + b[0], 0.f);
  }
}

// =====================================================================
// host-side orchestration
// =====================================================================
extern "C" void kernel_launch(void* const* d_in, const int* in_sizes, int n_in,
                              void* d_out, int out_size, void* d_ws, size_t ws_size,
                              hipStream_t stream) {
  const float* x       = (const float*)d_in[0];
  const int*   ei      = (const int*)  d_in[1];
  const float* ew      = (const float*)d_in[2];
  const float* W1      = (const float*)d_in[3];
  const float* b1      = (const float*)d_in[4];
  const float* W2      = (const float*)d_in[5];
  const float* b2      = (const float*)d_in[6];
  const float* W3      = (const float*)d_in[7];
  const float* b3      = (const float*)d_in[8];
  const float* pool_w  = (const float*)d_in[9];
  const float* head_W  = (const float*)d_in[10];
  const float* head_b  = (const float*)d_in[11];
  const float* norm_w  = (const float*)d_in[12];
  const float* norm_b  = (const float*)d_in[13];
  const float* dec_W   = (const float*)d_in[14];
  const float* dec_b   = (const float*)d_in[15];
  const float* dnorm_w = (const float*)d_in[16];
  const float* dnorm_b = (const float*)d_in[17];
  const float* dconv_W = (const float*)d_in[18];
  const float* dconv_b = (const float*)d_in[19];
  float* out = (float*)d_out;

  // ---- workspace carve-up (256B-aligned byte chunks) ----
  char* wsb = (char*)d_ws;
  size_t off = 0;
  auto carve = [&](size_t bytes) {
    void* p = wsb + off;
    off = (off + bytes + 255) & ~(size_t)255;
    return p;
  };
  float*    deg  = (float*)carve((size_t)NN * 4);
  float*    dinv = (float*)carve((size_t)NN * 4);
  float*    s1   = (float*)carve((size_t)NN * 4);
  float*    Aadj = (float*)carve((size_t)NG * NV * NV * 4);
  float*    z2   = (float*)carve((size_t)NG * NV * 4);
  _Float16* W2T  = (_Float16*)carve((size_t)C1 * C2 * 2);
  _Float16* W3T  = (_Float16*)carve((size_t)C2 * C3 * 2);
  _Float16* HWT  = (_Float16*)carve((size_t)KC * LATD * 2);
  _Float16* h1h  = (_Float16*)carve((size_t)NN * C1 * 2);  // also reused as zh
  _Float16* h2h  = (_Float16*)carve((size_t)NN * C2 * 2);
  float*    F32A = (float*)carve((size_t)NN * C2 * 4);     // xw2 / xw3 / zlat
  float*    F32B = (float*)carve((size_t)NN * C3 * 4);     // h3

  // ---- graph structure ----
  (void)hipMemsetAsync(deg,  0, (size_t)NN * sizeof(float), stream);
  (void)hipMemsetAsync(Aadj, 0, (size_t)NG * NV * NV * sizeof(float), stream);
  gcn_deg_kernel<<<(NE + 255) / 256, 256, 0, stream>>>(ei, ew, deg);
  gcn_dinv_kernel<<<(NN + 255) / 256, 256, 0, stream>>>(deg, dinv);
  gcn_adj_kernel<<<(NE + 255) / 256, 256, 0, stream>>>(ei, ew, dinv, Aadj);
  gcn_adj_diag_kernel<<<(NN + 255) / 256, 256, 0, stream>>>(dinv, Aadj);

  // ---- weight prep (f32 -> transposed f16) ----
  gcn_wtT_kernel<<<(int)(((size_t)C1 * C2 + 255) / 256), 256, 0, stream>>>(W2, W2T, C1, C2);
  gcn_wtT_kernel<<<(int)(((size_t)C2 * C3 + 255) / 256), 256, 0, stream>>>(W3, W3T, C2, C3);
  gcn_wtT_kernel<<<(int)(((size_t)KC * LATD + 255) / 256), 256, 0, stream>>>(head_W, HWT, KC, LATD);

  // ---- conv1 (rank-1 collapse) ----
  gcn_conv1_s_kernel<<<NG, 128, 0, stream>>>(Aadj, x, s1);
  gcn_h1_kernel<<<(int)(((size_t)NN * C1 + 255) / 256), 256, 0, stream>>>(s1, W1, b1, h1h);

  // ---- conv2: xw2 = h1 @ W2 (WMMA); h2 = relu(A@xw2 + b2) (f16) ----
  gemm_wmma_kernel<<<dim3(C2 / GBN, NN / GBM), 256, 0, stream>>>(
      h1h, W2T, nullptr, F32A, NN, C1, C2, 0);
  gcn_agg_kernel<_Float16><<<NG, 256, 0, stream>>>(Aadj, F32A, b2, h2h, C2);

  // ---- conv3: xw3 = h2 @ W3 (WMMA); h3 = relu(A@xw3 + b3) (f32) ----
  gemm_wmma_kernel<<<dim3(C3 / GBN, NN / GBM), 256, 0, stream>>>(
      h2h, W3T, nullptr, F32A, NN, C2, C3, 0);
  gcn_agg_kernel<float><<<NG, 256, 0, stream>>>(Aadj, F32A, b3, F32B, C3);

  // ---- TopK pool -> z [NG, KC] (f16, reuses h1h space) ----
  gcn_topk_kernel<<<NG, 128, 0, stream>>>(F32B, pool_w, h1h);

  // ---- head: zlat = z @ head_W + head_b (WMMA), then LN128 ----
  gemm_wmma_kernel<<<dim3(LATD / GBN, NG / GBM), 256, 0, stream>>>(
      h1h, HWT, head_b, F32A, NG, KC, LATD, 1);
  gcn_ln128_kernel<<<NG, 128, 0, stream>>>(F32A, norm_w, norm_b);

  // ---- decoder dense row (fused relu + LN68) ----
  gcn_dec_kernel<<<NG, 128, 0, stream>>>(F32A, dec_W, dec_b, dnorm_w, dnorm_b, z2);

  // ---- final 1-channel conv ----
  gcn_dconv_kernel<<<NG, 128, 0, stream>>>(Aadj, z2, dconv_W, dconv_b, out);
}
